// MultiHeadLatentAttention_68118181314720
// MI455X (gfx1250) — matrix operand order repaired and checked by reading, not compiled, not run on hardware
//
#include <hip/hip_runtime.h>
#include <cstddef>
#include <cstdint>

// ---------------------------------------------------------------------------
// MLA pipeline for gfx1250 (MI455X): bf16 WMMA GEMMs + transposed flash attn.
// Round 2: GEMM wave tile enlarged 32x32 -> 64x64 (16 FLOP/B -> 32.8 FLOP/B
// of L2 traffic) to keep the WMMA pipe fed from L2.
// ---------------------------------------------------------------------------

typedef __bf16 bf16_t;
typedef __attribute__((ext_vector_type(16))) __bf16       v16bf;
typedef __attribute__((ext_vector_type(8)))  float        v8f;
typedef __attribute__((ext_vector_type(4)))  float        f32x4;
typedef __attribute__((ext_vector_type(4)))  unsigned int u32x4;

union FragBf { v16bf v; u32x4 q[2]; };

__device__ __forceinline__ v8f vzero8() {
  v8f z = {0.f,0.f,0.f,0.f,0.f,0.f,0.f,0.f};
  return z;
}

__device__ __forceinline__ v8f wmma_bf16(v16bf a, v16bf b, v8f c) {
  // D = A(16x32 bf16) * B(32x16 bf16) + C(16x16 f32)
  return __builtin_amdgcn_wmma_f32_16x16x32_bf16(
      /*neg_a=*/false, a, /*neg_b=*/false, b,
      /*c_mod=*/(short)0, c, /*reuse_a=*/false, /*reuse_b=*/false);
}

// A fragment (16x32), source row-major, base points at (row0, k0).
// Layout: lanes 0-15 hold K[0..15] (V0..7, 2 per VGPR);
//         lanes 16-31 hold K[16..31].
__device__ __forceinline__ FragBf load_frag_a(const bf16_t* __restrict__ base,
                                              int lda, int l16, int h) {
  FragBf f;
  const bf16_t* p = base + (size_t)l16 * lda;
  f.q[0] = *(const u32x4*)(p + 16 * h);
  f.q[1] = *(const u32x4*)(p + 16 * h + 8);
  return f;
}

// B fragment (32x16), source column-major (Bt[n*ldb + k]), base at (k0, n0).
// Layout: lanes 0-15 hold K[0..15] (V0..7, 2 per VGPR); lanes 16-31 K[16..31].
__device__ __forceinline__ FragBf load_frag_b(const bf16_t* __restrict__ base,
                                              int ldb, int l16, int h) {
  FragBf f;
  const bf16_t* p = base + (size_t)l16 * ldb + 16 * h;
  f.q[0] = *(const u32x4*)(p);
  f.q[1] = *(const u32x4*)(p + 8);
  return f;
}

// ---------------------------------------------------------------------------
// Elementwise f32 -> bf16
// ---------------------------------------------------------------------------
__global__ __launch_bounds__(256) void to_bf16_kernel(
    const float* __restrict__ in, bf16_t* __restrict__ out, size_t n) {
  size_t i = (size_t)blockIdx.x * blockDim.x + threadIdx.x;
  if (i < n) out[i] = (bf16_t)in[i];
}

// W [K,N] f32 row-major  ->  Wt [N,K] bf16 (i.e. column-major B for the GEMM)
__global__ __launch_bounds__(256) void transpose_to_bf16_kernel(
    const float* __restrict__ W, bf16_t* __restrict__ Wt, int K, int N) {
  size_t i = (size_t)blockIdx.x * blockDim.x + threadIdx.x;
  size_t total = (size_t)K * N;
  if (i >= total) return;
  int n = (int)(i / K);
  int k = (int)(i - (size_t)n * K);
  Wt[i] = (bf16_t)W[(size_t)k * N + n];
}

// ---------------------------------------------------------------------------
// GEMM: out[M,N] = A[M,K](bf16,row-major) * Bt[N,K](bf16, B column-major)
// Block = 4 waves (2x2); wave computes 64x64 (4x4 WMMA tiles).
// Requires M%128==0, N%128==0, K%32==0 (all shapes here satisfy this).
// mode 0: f32 row-major; mode 1: bf16 row-major;
// mode 2: bf16 scattered into vt[b, h, d, s]  (head-transposed V)
// ---------------------------------------------------------------------------
__global__ __launch_bounds__(128) void gemm_bf16_kernel(
    const bf16_t* __restrict__ A, const bf16_t* __restrict__ Bt,
    void* __restrict__ out, int M, int N, int K, int mode, int S) {
  const int lane = threadIdx.x & 31;
  const int wave = threadIdx.x >> 5;  // 0..3
  const int l16 = lane & 15, h = lane >> 4;
  const int m0 = blockIdx.x * 128 + (wave >> 1) * 64;
  const int n0 = blockIdx.y * 128 + (wave & 1) * 64;

  v8f c[4][4];
#pragma unroll
  for (int i = 0; i < 4; ++i)
#pragma unroll
    for (int j = 0; j < 4; ++j) c[i][j] = vzero8();

  for (int k0 = 0; k0 < K; k0 += 32) {
    FragBf a[4], b[4];
#pragma unroll
    for (int i = 0; i < 4; ++i)
      a[i] = load_frag_a(A + (size_t)(m0 + 16 * i) * K + k0, K, l16, h);
#pragma unroll
    for (int j = 0; j < 4; ++j)
      b[j] = load_frag_b(Bt + (size_t)(n0 + 16 * j) * K + k0, K, l16, h);
#pragma unroll
    for (int i = 0; i < 4; ++i)
#pragma unroll
      for (int j = 0; j < 4; ++j)
        c[i][j] = wmma_bf16(a[i].v, b[j].v, c[i][j]);
  }

  // C layout: lane holds column n0+16j+l16; VGPR v holds row m0+16i+v+8h.
#pragma unroll
  for (int i = 0; i < 4; ++i) {
#pragma unroll
    for (int j = 0; j < 4; ++j) {
      const int n = n0 + 16 * j + l16;
#pragma unroll
      for (int v = 0; v < 8; ++v) {
        const int m = m0 + 16 * i + v + 8 * h;
        const float val = c[i][j][v];
        if (mode == 0) {
          ((float*)out)[(size_t)m * N + n] = val;
        } else if (mode == 1) {
          ((bf16_t*)out)[(size_t)m * N + n] = (bf16_t)val;
        } else {
          // vt[((b*16 + h_head)*128 + d) * S + s]
          const int bidx = m / S;
          const int s = m - bidx * S;
          const int hh = n >> 7;
          const int d = n & 127;
          ((bf16_t*)out)[((size_t)(bidx * 16 + hh) * 128 + d) * S + s] =
              (bf16_t)val;
        }
      }
    }
  }
}

// ---------------------------------------------------------------------------
// RoPE on f32 keys_pre -> bf16 kb.  Reference semantics (concat halves):
//   out[..., i]      = in[..., 2i]*cos - in[..., 2i+1]*sin   (i in [0,64))
//   out[..., 64 + i] = in[..., 2i]*sin + in[..., 2i+1]*cos
// per head (hd=128), angle = s * 10000^(-2i/128).
// ---------------------------------------------------------------------------
__global__ __launch_bounds__(256) void rope_to_bf16_kernel(
    const float* __restrict__ kp, bf16_t* __restrict__ kb, int S, int E,
    size_t total) {
  size_t idx = (size_t)blockIdx.x * blockDim.x + threadIdx.x;
  if (idx >= total) return;  // total = rows * E/2
  const int halfE = E >> 1;
  size_t row = idx / halfE;
  int ip = (int)(idx - row * halfE);
  int hh = ip >> 6;   // head
  int i = ip & 63;    // pair index within head
  int s = (int)(row % S);
  // 10000^(-2i/128) = exp(-(2i/128)*ln(10000))
  float freq = __expf(-(float)(2 * i) * (9.210340371976184f / 128.0f));
  float ang = (float)s * freq;
  float sn, cs;
  __sincosf(ang, &sn, &cs);
  const float* in = kp + row * E + (size_t)hh * 128;
  float e = in[2 * i];
  float o = in[2 * i + 1];
  bf16_t* ob = kb + row * E + (size_t)hh * 128;
  ob[i]      = (bf16_t)(e * cs - o * sn);
  ob[64 + i] = (bf16_t)(e * sn + o * cs);
}

// ---------------------------------------------------------------------------
// Flash attention (transposed form).  Per wave: one 16-row q tile of one
// (b,h).  S^T = K*Q^T via WMMA; softmax per q is per-lane + one shfl_xor(16);
// O^T = V^T * P^T accumulated over kv tiles of 32.
// ---------------------------------------------------------------------------
__global__ __launch_bounds__(256) void flash_attn_kernel(
    const bf16_t* __restrict__ qb, const bf16_t* __restrict__ kb,
    const bf16_t* __restrict__ vt, const float* __restrict__ mask,
    bf16_t* __restrict__ attn, int S) {
  const int E = 2048, HD = 128;
  const int lane = threadIdx.x & 31;
  const int wave = threadIdx.x >> 5;
  const int l16 = lane & 15, h = lane >> 4;
  const int bh = blockIdx.y;  // 0..31
  const int b = bh >> 4, hh = bh & 15;
  const int q0 = blockIdx.x * 128 + wave * 16;

  const size_t qrow = (size_t)b * S + q0;
  const bf16_t* Qbase = qb + qrow * E + (size_t)hh * HD;
  const bf16_t* Kbase = kb + (size_t)b * S * E + (size_t)hh * HD;
  const bf16_t* Vbase = vt + (size_t)bh * HD * S;  // [HD, S] row-major

  // Q used only as B-fragments (column-major view == Q row-major)
  FragBf bq[4];
#pragma unroll
  for (int kk = 0; kk < 4; ++kk)
    bq[kk] = load_frag_b(Qbase + 32 * kk, E, l16, h);

  v8f o[8];
#pragma unroll
  for (int t = 0; t < 8; ++t) o[t] = vzero8();
  float m_i = -1e30f, l_i = 0.f;
  const float scale = 0.08838834764831845f;  // 1/sqrt(128)

  for (int j0 = 0; j0 < S; j0 += 32) {
    v8f st0 = vzero8(), st1 = vzero8();
#pragma unroll
    for (int kk = 0; kk < 4; ++kk) {
      FragBf ak0 = load_frag_a(Kbase + (size_t)j0 * E + 32 * kk, E, l16, h);
      FragBf ak1 =
          load_frag_a(Kbase + (size_t)(j0 + 16) * E + 32 * kk, E, l16, h);
      st0 = wmma_bf16(ak0.v, bq[kk].v, st0);
      st1 = wmma_bf16(ak1.v, bq[kk].v, st1);
    }

    // Scale + mask.  Lane owns q column l16; its k indices are 8h+v (st0)
    // and 16+8h+v (st1), contiguous 8 -> vector mask loads.
    const float* mrow = mask + (size_t)(q0 + l16) * S + j0;
    f32x4 m0a = *(const f32x4*)(mrow + 8 * h);
    f32x4 m0b = *(const f32x4*)(mrow + 8 * h + 4);
    f32x4 m1a = *(const f32x4*)(mrow + 16 + 8 * h);
    f32x4 m1b = *(const f32x4*)(mrow + 16 + 8 * h + 4);

    float p0v[8], p1v[8];
#pragma unroll
    for (int v = 0; v < 8; ++v) {
      p0v[v] = st0[v] * scale + ((v < 4) ? m0a[v] : m0b[v - 4]);
      p1v[v] = st1[v] * scale + ((v < 4) ? m1a[v] : m1b[v - 4]);
    }

    // Online softmax (per-lane; halves combined via shfl_xor 16)
    float mx = -1e30f;
#pragma unroll
    for (int v = 0; v < 8; ++v) mx = fmaxf(mx, fmaxf(p0v[v], p1v[v]));
    mx = fmaxf(mx, __shfl_xor(mx, 16, 32));
    const float m_new = fmaxf(m_i, mx);
    float rsum = 0.f;
#pragma unroll
    for (int v = 0; v < 8; ++v) {
      p0v[v] = __expf(p0v[v] - m_new);
      p1v[v] = __expf(p1v[v] - m_new);
      rsum += p0v[v] + p1v[v];
    }
    rsum += __shfl_xor(rsum, 16, 32);
    const float alpha = __expf(m_i - m_new);
    l_i = l_i * alpha + rsum;
    m_i = m_new;
#pragma unroll
    for (int t = 0; t < 8; ++t)
#pragma unroll
      for (int v = 0; v < 8; ++v) o[t][v] *= alpha;

    // Repack P^T (two 16x16 C tiles) into a 32x16 bf16 B-fragment.
    float o0[8], o1[8];
#pragma unroll
    for (int v = 0; v < 8; ++v) {
      o0[v] = __shfl_xor(p0v[v], 16, 32);
      o1[v] = __shfl_xor(p1v[v], 16, 32);
    }
    FragBf pb;
#pragma unroll
    for (int v2 = 0; v2 < 4; ++v2) {  // B VGPR v2: K = 16h + 2*v2 (+1)
      float lo = h ? o1[2 * v2] : p0v[2 * v2];
      float hi = h ? o1[2 * v2 + 1] : p0v[2 * v2 + 1];
      pb.v[2 * v2] = (bf16_t)lo;
      pb.v[2 * v2 + 1] = (bf16_t)hi;
    }
#pragma unroll
    for (int v2 = 4; v2 < 8; ++v2) {  // K = 16h + 2*v2 (+1), i.e. 8..15 local
      int s2 = 2 * (v2 - 4);
      float lo = h ? p1v[s2] : o0[s2];
      float hi = h ? p1v[s2 + 1] : o0[s2 + 1];
      pb.v[2 * v2] = (bf16_t)lo;
      pb.v[2 * v2 + 1] = (bf16_t)hi;
    }

    // O^T += V^T(16d x 32k) * P^T(32k x 16q), 8 d-tiles covering hd=128
#pragma unroll
    for (int t = 0; t < 8; ++t) {
      FragBf av = load_frag_a(Vbase + (size_t)(t * 16) * S + j0, S, l16, h);
      o[t] = wmma_bf16(av.v, pb.v, o[t]);
    }
  }

  // Normalize and store O (row q0+l16, cols hh*128 + t*16 + 8h + v) as bf16.
  const float inv_l = 1.f / l_i;
  bf16_t* orow = attn + (qrow + l16) * E + (size_t)hh * HD;
#pragma unroll
  for (int t = 0; t < 8; ++t) {
    union { u32x4 q; bf16_t e[8]; } pk;
#pragma unroll
    for (int v = 0; v < 8; ++v) pk.e[v] = (bf16_t)(o[t][v] * inv_l);
    *(u32x4*)(orow + t * 16 + 8 * h) = pk.q;
  }
}

// ---------------------------------------------------------------------------
// Host side
// ---------------------------------------------------------------------------
extern "C" void kernel_launch(void* const* d_in, const int* in_sizes, int n_in,
                              void* d_out, int out_size, void* d_ws,
                              size_t ws_size, hipStream_t stream) {
  (void)in_sizes; (void)n_in; (void)out_size; (void)ws_size;
  const int B = 2, S = 2048, E = 2048, C = 512;
  const int M = B * S;  // 4096 rows

  const float* x      = (const float*)d_in[0];
  const float* mask   = (const float*)d_in[1];
  const float* W_dkv  = (const float*)d_in[2];
  const float* W_uv   = (const float*)d_in[3];
  const float* W_kr   = (const float*)d_in[4];
  const float* W_dq   = (const float*)d_in[5];
  const float* W_uq   = (const float*)d_in[6];
  const float* W_o    = (const float*)d_in[7];
  float* out = (float*)d_out;

  char* ws = (char*)d_ws;
  size_t off = 0;
  auto alloc = [&](size_t bytes) {
    size_t o = off;
    off += (bytes + 255) & ~(size_t)255;
    return o;
  };
  bf16_t* xb     = (bf16_t*)(ws + alloc((size_t)M * E * 2));
  bf16_t* wt_dkv = (bf16_t*)(ws + alloc((size_t)E * C * 2));
  bf16_t* wt_uv  = (bf16_t*)(ws + alloc((size_t)C * E * 2));
  bf16_t* wt_kr  = (bf16_t*)(ws + alloc((size_t)E * E * 2));
  bf16_t* wt_dq  = (bf16_t*)(ws + alloc((size_t)E * C * 2));
  bf16_t* wt_uq  = (bf16_t*)(ws + alloc((size_t)C * E * 2));
  bf16_t* wt_o   = (bf16_t*)(ws + alloc((size_t)E * E * 2));
  bf16_t* kvc    = (bf16_t*)(ws + alloc((size_t)M * C * 2));
  bf16_t* vt     = (bf16_t*)(ws + alloc((size_t)M * E * 2));
  float*  kpre   = (float*) (ws + alloc((size_t)M * E * 4));
  bf16_t* kb     = (bf16_t*)(ws + alloc((size_t)M * E * 2));
  bf16_t* qc     = (bf16_t*)(ws + alloc((size_t)M * C * 2));
  bf16_t* qb     = (bf16_t*)(ws + alloc((size_t)M * E * 2));
  bf16_t* attn   = (bf16_t*)(ws + alloc((size_t)M * E * 2));

  auto blocks = [](size_t n) { return (unsigned)((n + 255) / 256); };

  // 1) x -> bf16
  size_t nx = (size_t)M * E;
  to_bf16_kernel<<<blocks(nx), 256, 0, stream>>>(x, xb, nx);

  // 2) weight transposes (f32 -> bf16 column-major-B)
  transpose_to_bf16_kernel<<<blocks((size_t)E * C), 256, 0, stream>>>(W_dkv, wt_dkv, E, C);
  transpose_to_bf16_kernel<<<blocks((size_t)C * E), 256, 0, stream>>>(W_uv,  wt_uv,  C, E);
  transpose_to_bf16_kernel<<<blocks((size_t)E * E), 256, 0, stream>>>(W_kr,  wt_kr,  E, E);
  transpose_to_bf16_kernel<<<blocks((size_t)E * C), 256, 0, stream>>>(W_dq,  wt_dq,  E, C);
  transpose_to_bf16_kernel<<<blocks((size_t)C * E), 256, 0, stream>>>(W_uq,  wt_uq,  C, E);
  transpose_to_bf16_kernel<<<blocks((size_t)E * E), 256, 0, stream>>>(W_o,   wt_o,   E, E);

  // 3) kv_c = x @ W_dkv              [4096,512]  -> bf16
  gemm_bf16_kernel<<<dim3(M / 128, C / 128), 128, 0, stream>>>(xb, wt_dkv, kvc, M, C, E, 1, S);
  // 4) values = kv_c @ W_uv          -> vt[b,h,d,s] bf16 (transposed store)
  gemm_bf16_kernel<<<dim3(M / 128, E / 128), 128, 0, stream>>>(kvc, wt_uv, vt, M, E, C, 2, S);
  // 5) keys_pre = x @ W_kr           [4096,2048] f32
  gemm_bf16_kernel<<<dim3(M / 128, E / 128), 128, 0, stream>>>(xb, wt_kr, kpre, M, E, E, 0, S);
  // 6) RoPE -> kb bf16
  size_t nrope = (size_t)M * (E / 2);
  rope_to_bf16_kernel<<<blocks(nrope), 256, 0, stream>>>(kpre, kb, S, E, nrope);
  // 7) q_c = x @ W_dq                [4096,512] bf16
  gemm_bf16_kernel<<<dim3(M / 128, C / 128), 128, 0, stream>>>(xb, wt_dq, qc, M, C, E, 1, S);
  // 8) queries = q_c @ W_uq          [4096,2048] bf16
  gemm_bf16_kernel<<<dim3(M / 128, E / 128), 128, 0, stream>>>(qc, wt_uq, qb, M, E, C, 1, S);
  // 9) attention -> attn bf16
  flash_attn_kernel<<<dim3(S / 128, B * 16), 256, 0, stream>>>(qb, kb, vt, mask, attn, S);
  // 10) out = attn @ W_o             f32 -> d_out
  gemm_bf16_kernel<<<dim3(M / 128, E / 128), 128, 0, stream>>>(attn, wt_o, out, M, E, E, 0, S);
}
